// ODConv1d_32384053411955
// MI455X (gfx1250) — matrix operand order repaired and checked by reading, not compiled, not run on hardware
//
#include <hip/hip_runtime.h>
#include <math.h>

#define B_  32
#define L_  4096
#define C_  256
#define O_  256
#define K_  3
#define KN_ 4
#define A_  16
#define CK_ (C_ * K_)        // 768
#define BN_EPS 1e-5f

typedef __attribute__((ext_vector_type(16))) _Float16 v16h;
typedef __attribute__((ext_vector_type(8)))  float    v8f;
typedef unsigned int u32x4 __attribute__((ext_vector_type(4)));
typedef int          i32x4 __attribute__((ext_vector_type(4)));
typedef int          i32x8 __attribute__((ext_vector_type(8)));

union FragU { unsigned int u[8]; v16h h; };
union PackU { _Float16 h[2]; unsigned int u; };

// ---------------------------------------------------------------------------
// Zero the pooled-sum accumulator region of the workspace.
// ---------------------------------------------------------------------------
__global__ void zero_ws_kernel(float* __restrict__ pooled) {
    int i = blockIdx.x * 256 + threadIdx.x;
    if (i < B_ * C_) pooled[i] = 0.0f;
}

// ---------------------------------------------------------------------------
// Global-average pool over L: pooled_sum[b][c] += sum of a 512-row chunk.
// ---------------------------------------------------------------------------
__global__ void pool_kernel(const float* __restrict__ x,
                            float* __restrict__ pooled) {
    const int b     = blockIdx.y;
    const int chunk = blockIdx.x;        // 0..7, each covers 512 rows
    const int c     = threadIdx.x;       // 0..255
    const float* xp = x + ((size_t)b * L_ + (size_t)chunk * 512) * C_ + c;
    float s = 0.0f;
#pragma unroll 4
    for (int l = 0; l < 512; ++l) s += xp[(size_t)l * C_];
    atomicAdd(&pooled[b * C_ + c], s);
}

// ---------------------------------------------------------------------------
// Attention head: h = relu(BN(pooled @ w_fc^T)); four small heads.
// ---------------------------------------------------------------------------
__global__ void attn_kernel(const float* __restrict__ pooled_sum,
                            const float* __restrict__ w_fc,
                            const float* __restrict__ bn_gamma,
                            const float* __restrict__ bn_beta,
                            const float* __restrict__ bn_mean,
                            const float* __restrict__ bn_var,
                            const float* __restrict__ w_ch,  const float* __restrict__ b_ch,
                            const float* __restrict__ w_fil, const float* __restrict__ b_fil,
                            const float* __restrict__ w_sp,  const float* __restrict__ b_sp,
                            const float* __restrict__ w_ker, const float* __restrict__ b_ker,
                            float* __restrict__ ch_att, float* __restrict__ fil_att,
                            float* __restrict__ sp_att, float* __restrict__ ker_att) {
    __shared__ float h[B_][A_];
    const int t = threadIdx.x;           // 0..511

    {   // h[b][a]
        const int b = t >> 4, a = t & 15;
        const float* p = pooled_sum + b * C_;
        const float* w = w_fc + a * C_;
        float acc = 0.0f;
        for (int c = 0; c < C_; ++c) acc += p[c] * w[c];
        acc *= (1.0f / (float)L_);                       // mean
        acc = (acc - bn_mean[a]) * rsqrtf(bn_var[a] + BN_EPS) * bn_gamma[a] + bn_beta[a];
        h[b][a] = fmaxf(acc, 0.0f);
    }
    __syncthreads();

    for (int q = 0; q < 16; ++q) {       // channel attention [B, C]
        int idx = t + 512 * q;
        int b = idx >> 8, i = idx & 255;
        float acc = b_ch[i];
        for (int a = 0; a < A_; ++a) acc += h[b][a] * w_ch[i * A_ + a];
        ch_att[idx] = 1.0f / (1.0f + __expf(-acc));
    }
    for (int q = 0; q < 16; ++q) {       // filter attention [B, O]
        int idx = t + 512 * q;
        int b = idx >> 8, o = idx & 255;
        float acc = b_fil[o];
        for (int a = 0; a < A_; ++a) acc += h[b][a] * w_fil[o * A_ + a];
        fil_att[idx] = 1.0f / (1.0f + __expf(-acc));
    }
    if (t < B_ * K_) {                   // spatial attention [B, K]
        int b = t / K_, k = t % K_;
        float acc = b_sp[k];
        for (int a = 0; a < A_; ++a) acc += h[b][a] * w_sp[k * A_ + a];
        sp_att[t] = 1.0f / (1.0f + __expf(-acc));
    }
    if (t < B_) {                        // kernel attention softmax [B, KN]
        float z[KN_], m = -1e30f;
        for (int n = 0; n < KN_; ++n) {
            float acc = b_ker[n];
            for (int a = 0; a < A_; ++a) acc += h[t][a] * w_ker[n * A_ + a];
            z[n] = acc; m = fmaxf(m, acc);
        }
        float s = 0.0f;
        for (int n = 0; n < KN_; ++n) { z[n] = __expf(z[n] - m); s += z[n]; }
        for (int n = 0; n < KN_; ++n) ker_att[t * KN_ + n] = z[n] / s;
    }
}

// ---------------------------------------------------------------------------
// Build per-sample effective weight, folding ALL four attentions:
//   W_eff[b][o][k*C+i] = fil[b,o]*ch[b,i]*sp[b,k]*sum_n ker[b,n]*weight[n,o,i,k]
// Stored as packed f16 pairs (along ck) ready for WMMA A-fragments.
// ---------------------------------------------------------------------------
__global__ void build_weff_kernel(const float* __restrict__ weight,
                                  const float* __restrict__ ch_att,
                                  const float* __restrict__ fil_att,
                                  const float* __restrict__ sp_att,
                                  const float* __restrict__ ker_att,
                                  unsigned int* __restrict__ Wps) {
    const int o = blockIdx.x;
    const int b = blockIdx.y;
    float kw[KN_];
#pragma unroll
    for (int n = 0; n < KN_; ++n) kw[n] = ker_att[b * KN_ + n];
    const float f = fil_att[b * O_ + o];
    unsigned int* outp = Wps + ((size_t)b * O_ + o) * (CK_ / 2);

    for (int p = threadIdx.x; p < CK_ / 2; p += blockDim.x) {
        const int ck = 2 * p;
        const int k  = ck >> 8;          // / C_
        const int i  = ck & 255;         // % C_ (even)
        const float* wbase = weight + (size_t)o * C_ * K_ + (size_t)i * K_ + k;
        float a0 = 0.0f, a1 = 0.0f;
#pragma unroll
        for (int n = 0; n < KN_; ++n) {
            a0 += kw[n] * wbase[(size_t)n * O_ * C_ * K_];
            a1 += kw[n] * wbase[(size_t)n * O_ * C_ * K_ + K_];
        }
        const float scale = f * sp_att[b * K_ + k];
        PackU pk;
        pk.h[0] = (_Float16)(scale * ch_att[b * C_ + i]     * a0);
        pk.h[1] = (_Float16)(scale * ch_att[b * C_ + i + 1] * a1);
        outp[p] = pk.u;
    }
}

// ---------------------------------------------------------------------------
// Main conv as per-batch GEMM with v_wmma_f32_16x16x32_f16.
//   out[b] (O x L) = W_eff[b] (O x CK) * im2col(x[b]) (CK x L)
// A tile (128 rows x 64 B, row stride 1536 B) is fetched by the Tensor Data
// Mover straight into LDS; TDM padding (16 dwords data + 4 dwords pad) gives a
// 20-dword row stride, which is bank-conflict-free AND 16B-aligns every
// fragment group so fragment loads are ds_load_b128 (2 per fragment).
// B tile staged synchronously (f32->f16) in fragment order [l][ck-pair] while
// the TDM transfer is in flight.
// ---------------------------------------------------------------------------
#define TM 128
#define TN 128
#define LDS_STRIDE 20     // dwords; TDM pad 16+4; 20*m mod 64 distinct, 16B-aligned
#define A_LDS_DW (TM * LDS_STRIDE)           // 2560 dwords
#define B_LDS_DW (TN * LDS_STRIDE)           // 2560 dwords

__global__ __launch_bounds__(256)
void conv_wmma_kernel(const float* __restrict__ x,
                      const unsigned int* __restrict__ Wps,
                      float* __restrict__ out) {
    __shared__ unsigned int lds_all[A_LDS_DW + B_LDS_DW];
    unsigned int* a_lds = lds_all;                 // LDS byte offset 0 (for D#)
    unsigned int* b_lds = lds_all + A_LDS_DW;      // [l][ck-pair], stride 20

    const int b  = blockIdx.z;
    const int o0 = blockIdx.y * TM;
    const int l0 = blockIdx.x * TN;

    const int t    = threadIdx.x;
    const int lane = t & 31;
    const int wave = t >> 5;
    const int wm = wave & 1, wn = wave >> 1;
    const int mw = wm * 64, nw = wn * 32;
    const int half = lane >> 4;
    const int lm   = lane & 15;

    v8f zero = {};
    v8f acc[4][2];
#pragma unroll
    for (int s = 0; s < 4; ++s)
#pragma unroll
        for (int u = 0; u < 2; ++u) acc[s][u] = zero;

    const unsigned int* wrow = Wps + ((size_t)b * O_ + o0) * (CK_ / 2);

    // TDM D# group1 (constant): data_size=4B, pad_enable, pad_interval=16dw,
    // pad_amount=4dw (-> 20dw LDS row stride), tensor_dim0=384 dw,
    // tensor_dim1=256 rows, tile 16dw x 128 rows, row stride 384 dw.
    i32x8 tdm_g1;
    tdm_g1[0] = (2 << 16) | (1 << 20) | (3 << 22) | (3 << 25);
    tdm_g1[1] = (int)(384u << 16);     // tensor_dim0[15:0] in bits 63:48
    tdm_g1[2] = (int)(256u << 16);     // tensor_dim1[15:0] in bits 95:80
    tdm_g1[3] = (int)(16u << 16);      // tile_dim0 in bits 127:112
    tdm_g1[4] = 128;                   // tile_dim1 in bits 143:128
    tdm_g1[5] = 384;                   // tensor_dim0_stride[31:0]
    tdm_g1[6] = 0;
    tdm_g1[7] = 0;
    i32x4 tdm_zero4 = {0, 0, 0, 0};

    // B-staging map: ip = channel-pair 0..15, lb = base l 0..15 (x8 strided)
    const int ip = t & 15;
    const int lb = t >> 4;

    for (int ck0 = 0; ck0 < CK_; ck0 += 32) {
        __syncthreads();
        // ---- A tile via Tensor Data Mover (wave 0 issues the DMA) ----
        if (wave == 0) {
            unsigned long long gaddr =
                (unsigned long long)(const void*)(wrow + (ck0 >> 1));
            u32x4 g0;
            g0[0] = 1u;                                  // count=1, user mode
            g0[1] = 0u;                                  // lds_addr = a_lds @ 0
            g0[2] = (unsigned)(gaddr & 0xFFFFFFFFull);
            g0[3] = (unsigned)((gaddr >> 32) & 0x01FFFFFFull) | (2u << 30); // type=2
#if __clang_major__ >= 23
            __builtin_amdgcn_tensor_load_to_lds(g0, tdm_g1, tdm_zero4, tdm_zero4,
                                                tdm_g1, 0);
#else
            __builtin_amdgcn_tensor_load_to_lds(g0, tdm_g1, tdm_zero4, tdm_zero4, 0);
#endif
        }
        // ---- stage B tile in fragment order: b_lds[l][ck-pair] ----
        {
            const int kk = ck0 >> 8;          // conv tap for this 32-chunk
            const int i0 = ck0 & 255;
#pragma unroll
            for (int q = 0; q < 8; ++q) {
                const int ll = lb + 16 * q;               // 0..127
                const int lg = l0 + ll + kk - 1;          // padded position
                unsigned int pack = 0u;
                if (lg >= 0 && lg < L_) {
                    const float2* xp = reinterpret_cast<const float2*>(
                        x + ((size_t)b * L_ + lg) * C_ + i0 + 2 * ip);
                    float2 f = *xp;
                    PackU pk;
                    pk.h[0] = (_Float16)f.x;
                    pk.h[1] = (_Float16)f.y;
                    pack = pk.u;
                }
                b_lds[ll * LDS_STRIDE + ip] = pack;
            }
        }
        if (wave == 0) {
            __builtin_amdgcn_s_wait_tensorcnt(0);         // A tile landed
        }
        __syncthreads();

        // ---- fragments: 2x ds_load_b128 each, per ISA 16-bit layouts ----
        FragU afr[4], bfr[2];
#pragma unroll
        for (int s = 0; s < 4; ++s) {
            const int m = mw + s * 16 + lm;
            // A frag j=0..3 -> K-pairs half*4+0..3 ; j=4..7 -> 8+half*4+0..3
            const uint4* ap = reinterpret_cast<const uint4*>(
                &a_lds[m * LDS_STRIDE + half * 4]);
            uint4 lo = ap[0];
            uint4 hi = ap[2];                 // +8 dwords
            afr[s].u[0] = lo.x; afr[s].u[1] = lo.y;
            afr[s].u[2] = lo.z; afr[s].u[3] = lo.w;
            afr[s].u[4] = hi.x; afr[s].u[5] = hi.y;
            afr[s].u[6] = hi.z; afr[s].u[7] = hi.w;
        }
#pragma unroll
        for (int u = 0; u < 2; ++u) {
            const int n = nw + u * 16 + lm;
            // B frag j=0..3 -> K-pairs half*4+0..3 ; j=4..7 -> 8+half*4+0..3
            const uint4* bp = reinterpret_cast<const uint4*>(
                &b_lds[n * LDS_STRIDE + half * 4]);
            uint4 lo = bp[0];
            uint4 hi = bp[2];                 // +8 dwords
            bfr[u].u[0] = lo.x; bfr[u].u[1] = lo.y;
            bfr[u].u[2] = lo.z; bfr[u].u[3] = lo.w;
            bfr[u].u[4] = hi.x; bfr[u].u[5] = hi.y;
            bfr[u].u[6] = hi.z; bfr[u].u[7] = hi.w;
        }
#pragma unroll
        for (int s = 0; s < 4; ++s)
#pragma unroll
            for (int u = 0; u < 2; ++u)
                acc[s][u] = __builtin_amdgcn_wmma_f32_16x16x32_f16(
                    false, afr[s].h, false, bfr[u].h,
                    (short)0, acc[s][u], false, false);
    }

    // ---- epilogue: C layout is M = r + 8*half, N = lane&15 ----
#pragma unroll
    for (int s = 0; s < 4; ++s) {
        const int obase = o0 + mw + s * 16 + half * 8;
#pragma unroll
        for (int u = 0; u < 2; ++u) {
            const int l = l0 + nw + u * 16 + lm;
            float* op = out + ((size_t)b * O_ + obase) * L_ + l;
#pragma unroll
            for (int r = 0; r < 8; ++r)
                op[(size_t)r * L_] = acc[s][u][r];
        }
    }
}

// ---------------------------------------------------------------------------
// Launcher
// ---------------------------------------------------------------------------
extern "C" void kernel_launch(void* const* d_in, const int* in_sizes, int n_in,
                              void* d_out, int out_size, void* d_ws, size_t ws_size,
                              hipStream_t stream) {
    const float* x      = (const float*)d_in[0];
    const float* weight = (const float*)d_in[1];
    const float* w_fc   = (const float*)d_in[2];
    const float* gamma  = (const float*)d_in[3];
    const float* beta   = (const float*)d_in[4];
    const float* mean   = (const float*)d_in[5];
    const float* var    = (const float*)d_in[6];
    const float* w_ch   = (const float*)d_in[7];
    const float* b_ch   = (const float*)d_in[8];
    const float* w_fil  = (const float*)d_in[9];
    const float* b_fil  = (const float*)d_in[10];
    const float* w_sp   = (const float*)d_in[11];
    const float* b_sp   = (const float*)d_in[12];
    const float* w_ker  = (const float*)d_in[13];
    const float* b_ker  = (const float*)d_in[14];
    float* out = (float*)d_out;

    // workspace layout (floats)
    float* ws      = (float*)d_ws;
    float* pooled  = ws;                         // 8192
    float* ch_att  = pooled + B_ * C_;           // 8192
    float* fil_att = ch_att + B_ * C_;           // 8192
    float* sp_att  = fil_att + B_ * O_;          // 96 (pad 128)
    float* ker_att = sp_att + 128;               // 128
    unsigned int* Wps = (unsigned int*)(ker_att + 128);  // B*O*(CK/2) u32 ~ 12 MB

    zero_ws_kernel<<<(B_ * C_ + 255) / 256, 256, 0, stream>>>(pooled);
    pool_kernel<<<dim3(8, B_), 256, 0, stream>>>(x, pooled);
    attn_kernel<<<1, 512, 0, stream>>>(pooled, w_fc, gamma, beta, mean, var,
                                       w_ch, b_ch, w_fil, b_fil,
                                       w_sp, b_sp, w_ker, b_ker,
                                       ch_att, fil_att, sp_att, ker_att);
    build_weff_kernel<<<dim3(O_, B_), 128, 0, stream>>>(weight, ch_att, fil_att,
                                                        sp_att, ker_att, Wps);
    conv_wmma_kernel<<<dim3(L_ / TN, O_ / TM, B_), 256, 0, stream>>>(x, Wps, out);
    (void)in_sizes; (void)n_in; (void)out_size; (void)ws_size;
}
